// DocumentLevelLstmWithMotPooling_53420803228388
// MI455X (gfx1250) — matrix operand order repaired and checked by reading, not compiled, not run on hardware
//
#include <hip/hip_runtime.h>
#include <hip/hip_bf16.h>

typedef __bf16 bf16_t;
typedef __attribute__((ext_vector_type(16))) __bf16 v16bf;
typedef __attribute__((ext_vector_type(8)))  __bf16 v8bf;
typedef __attribute__((ext_vector_type(8)))  float  v8f;

// Problem constants
#define BATCH 64
#define TLEN  1024
#define DIN   300
#define DPAD  320          // DIN padded to multiple of 32
#define HDIM  512
#define GDIM  2048         // 4*H
#define NTILES 128         // GDIM/16
#define KT_X  10           // DPAD/32
#define KT_H  16           // HDIM/32
#define MROWS (BATCH*TLEN) // 65536
#define NBLK_SCAN 32       // one workgroup per 16-col slice of h

__device__ __forceinline__ float sigm(float x) { return 1.0f / (1.0f + __expf(-x)); }

__device__ __forceinline__ v8f wmma_bf16(v16bf a, v16bf b, v8f c) {
  // D = A(16x32) * B(32x16) + C, fp32 accumulate
  return __builtin_amdgcn_wmma_f32_16x16x32_bf16(false, a, false, b, (short)0, c, false, false);
}

// A fragment (16x32 bf16) loaded straight from a row-major matrix.
// Lane l<16 : row m0+l,      halves K[k0+0..7]  and K[k0+16..23]
// Lane l>=16: row m0+(l-16), halves K[k0+8..15] and K[k0+24..31]
__device__ __forceinline__ v16bf load_a_frag(const bf16_t* __restrict__ A, int lda,
                                             int m0, int k0, int lane) {
  int row    = m0 + (lane & 15);
  int kfirst = k0 + ((lane >> 4) << 3);
  const bf16_t* p = A + (size_t)row * lda + kfirst;
  v8bf lo = *(const v8bf*)(p);
  v8bf hi = *(const v8bf*)(p + 16);
  return __builtin_shufflevector(lo, hi, 0,1,2,3,4,5,6,7,8,9,10,11,12,13,14,15);
}

// B fragment from a packed tile buffer (global): tile = 512 bf16; lane l reads
// chunk l (first half) and chunk l+32 (second half), each 8 bf16 contiguous.
__device__ __forceinline__ v16bf load_b_frag(const bf16_t* __restrict__ Bp,
                                             size_t tileIdx, int lane) {
  const bf16_t* base = Bp + tileIdx * 512;
  v8bf lo = *(const v8bf*)(base + (size_t)lane * 8);
  v8bf hi = *(const v8bf*)(base + ((size_t)lane + 32) * 8);
  return __builtin_shufflevector(lo, hi, 0,1,2,3,4,5,6,7,8,9,10,11,12,13,14,15);
}

// Same layout, but tiles resident in LDS -> ds_load_b128 x2
__device__ __forceinline__ v16bf load_b_frag_lds(const bf16_t* base, int tileIdx, int lane) {
  const bf16_t* p = base + (size_t)tileIdx * 512;
  v8bf lo = *(const v8bf*)(p + lane * 8);
  v8bf hi = *(const v8bf*)(p + (lane + 32) * 8);
  return __builtin_shufflevector(lo, hi, 0,1,2,3,4,5,6,7,8,9,10,11,12,13,14,15);
}

// ---------------- init: zero h ping buffer + barrier counter ----------------
__global__ void init_kernel(bf16_t* h0, unsigned* bar) {
  int i = blockIdx.x * 256 + threadIdx.x;
  if (i < BATCH * HDIM) h0[i] = (bf16_t)0.0f;
  if (i == 0) *bar = 0u;
}

// ---------------- pack essays -> bf16, K padded to 320 ----------------
__global__ void pack_essays_kernel(const float* __restrict__ essays, bf16_t* __restrict__ Ap) {
  size_t i = (size_t)blockIdx.x * 256 + threadIdx.x; // over MROWS*DPAD
  if (i >= (size_t)MROWS * DPAD) return;
  size_t m = i / DPAD;
  int    k = (int)(i % DPAD);
  Ap[i] = (k < DIN) ? (bf16_t)essays[m * DIN + k] : (bf16_t)0.0f;
}

// ---------------- pack W_lstm into WMMA-B tile order ----------------
// chunk c of tile (kt,nt): l=c&31, half=c>>5; lane-l column N=nt*16+(l&15),
// K rows kt*32 + (l&16 ? 8:0) + half*16 + [0..7]
__global__ void pack_weights_kernel(const float* __restrict__ W,
                                    bf16_t* __restrict__ BxP, bf16_t* __restrict__ BhP) {
  int tid = blockIdx.x * 256 + threadIdx.x;
  const int total_x = KT_X * NTILES * 64;
  const int total_h = KT_H * NTILES * 64;
  if (tid >= total_x + total_h) return;
  bool isX = tid < total_x;
  int rel = isX ? tid : tid - total_x;
  int KT  = isX ? KT_X : KT_H;
  int tile = rel >> 6;
  int c    = rel & 63;
  int nt = tile / KT, kt = tile % KT;        // tileIdx = nt*KT + kt
  int l = c & 31, half = c >> 5;
  int n  = nt * 16 + (l & 15);
  int kk = kt * 32 + ((l & 16) ? 8 : 0) + half * 16;
  bf16_t* out = (isX ? BxP : BhP) + (size_t)tile * 512 + (size_t)c * 8;
#pragma unroll
  for (int i = 0; i < 8; ++i) {
    int krow = kk + i;
    float v = 0.0f;
    if (isX) { if (krow < DIN) v = W[(size_t)krow * GDIM + n]; }
    else     {                 v = W[(size_t)(DIN + krow) * GDIM + n]; }
    out[i] = (bf16_t)v;
  }
}

// ---------------- x_proj GEMM: (65536x320)x(320x2048) + bias -> (T,B,G) fp32 ----------------
__global__ __launch_bounds__(128) void xproj_gemm_kernel(
    const bf16_t* __restrict__ Ap, const bf16_t* __restrict__ BxP,
    const float* __restrict__ b_lstm, float* __restrict__ xproj) {
  const int lane = threadIdx.x & 31;
  const int wave = threadIdx.x >> 5;
  const int mt = blockIdx.y;                 // 0..4095
  const int nt = blockIdx.x * 4 + wave;      // 0..127
  const int m0 = mt * 16, n0 = nt * 16;
  v8f acc = {};
#pragma unroll
  for (int kt = 0; kt < KT_X; ++kt) {
    v16bf a = load_a_frag(Ap, DPAD, m0, kt * 32, lane);
    v16bf b = load_b_frag(BxP, (size_t)nt * KT_X + kt, lane);
    acc = wmma_bf16(a, b, acc);
  }
  // C layout: vgpr r, lane l -> row r + (l<16?0:8), col l&15.  m = b*T + t.
  const int bidx = m0 >> 10;
  const int t0   = m0 & (TLEN - 1);
  const int col  = n0 + (lane & 15);
  const float bias = b_lstm[col];
  const int rshift = (lane >> 4) << 3;
#pragma unroll
  for (int r = 0; r < 8; ++r) {
    int t = t0 + r + rshift;
    // 512 MiB streamed once, read once much later: bypass-friendly NT store
    __builtin_nontemporal_store(acc[r] + bias,
        xproj + ((size_t)t * BATCH + bidx) * GDIM + col);
  }
}

// ---------------- persistent LSTM scan ----------------
// 32 workgroups x 4 waves. Wave owns h-tile (m0 = wave*16 rows, n0 = wg*16 cols);
// computes 4 gate tiles (i,j,f,o at col offsets 0/512/1024/1536). Cell state c
// and pooling sum stay in registers for all 1024 steps. The block's 64 KB slice
// of packed Wh is preloaded into LDS once (per-step B-frags come from ds_load).
// Next step's x_proj slab is prefetched into cache during compute. Cross-step
// sync via monotonic atomic counter barrier + ping-pong h buffers.
__global__ __launch_bounds__(128) void lstm_scan_kernel(
    const float* __restrict__ xproj, const bf16_t* __restrict__ BhP,
    bf16_t* __restrict__ h0, bf16_t* __restrict__ h1,
    float* __restrict__ hsum, unsigned* __restrict__ bar) {
  const int lane = threadIdx.x & 31;
  const int wave = threadIdx.x >> 5;
  const int wg   = blockIdx.x;
  const int m0 = wave * 16;
  const int n0 = wg * 16;
  const int rshift = (lane >> 4) << 3;
  const int col = n0 + (lane & 15);

  // Preload this block's Wh slice: 4 gates x 16 k-tiles x 1 KB = 64 KB in LDS.
  __shared__ __align__(16) bf16_t ldsWh[4 * KT_H * 512];
  for (int j = threadIdx.x; j < 4 * KT_H * 64; j += 128) {  // 16B chunks
    int tile = j >> 6, c = j & 63;
    int g = tile >> 4, kt = tile & 15;
    const v8bf* src = (const v8bf*)(BhP +
        ((size_t)(g * 32 + wg) * KT_H + kt) * 512 + (size_t)c * 8);
    *(v8bf*)(ldsWh + (size_t)tile * 512 + (size_t)c * 8) = *src;
  }
  __syncthreads();

  v8f cstate = {};
  v8f hs = {};

  for (int t = 0; t < TLEN; ++t) {
    const bf16_t* hin  = (t & 1) ? h1 : h0;
    bf16_t*       hout = (t & 1) ? h0 : h1;

    // init gate accumulators from x_proj (already bias-added)
    const float* xp = xproj + (size_t)t * BATCH * GDIM;
    v8f acc[4];
#pragma unroll
    for (int g = 0; g < 4; ++g) {
#pragma unroll
      for (int r = 0; r < 8; ++r) {
        int row = m0 + r + rshift;
        acc[g][r] = xp[(size_t)row * GDIM + g * HDIM + col];
      }
    }
    // gates += h_prev @ Wh   (K = 512, 16 k-steps, 4 interleaved WMMA chains;
    // B operands from LDS, A operands from the 64 KB L2-resident h buffer)
#pragma unroll 4
    for (int kt = 0; kt < KT_H; ++kt) {
      v16bf a = load_a_frag(hin, HDIM, m0, kt * 32, lane);
#pragma unroll
      for (int g = 0; g < 4; ++g) {
        v16bf b = load_b_frag_lds(ldsWh, g * 16 + kt, lane);
        acc[g] = wmma_bf16(a, b, acc[g]);
      }
    }
    // prefetch next step's x_proj region for this wave (hides HBM latency
    // behind the elementwise work + barrier): 64 lines = 2 per lane
    if (t + 1 < TLEN) {
      const float* nxt = xproj + (size_t)(t + 1) * BATCH * GDIM;
#pragma unroll
      for (int q = 0; q < 2; ++q) {
        int idx = lane + 32 * q;            // 0..63 -> (gate, row)
        int g = idx >> 4, row = m0 + (idx & 15);
        __builtin_prefetch(nxt + (size_t)row * GDIM + g * HDIM + n0, 0, 1);
      }
    }
    // elementwise LSTM cell (fp32), keep c & pooled sum in registers
#pragma unroll
    for (int r = 0; r < 8; ++r) {
      float iv = acc[0][r], jv = acc[1][r], fv = acc[2][r], ov = acc[3][r];
      float cf = cstate[r] * sigm(fv + 1.0f) + sigm(iv) * tanhf(jv);
      float hf = tanhf(cf) * sigm(ov);
      cstate[r] = cf;
      hs[r] += hf;
      int row = m0 + r + rshift;
      hout[(size_t)row * HDIM + col] = (bf16_t)hf;
    }
    // device-wide barrier: every thread releases its h stores, one thread
    // arrives + spins, every thread acquires afterwards.
    __threadfence();
    __syncthreads();
    if (threadIdx.x == 0) {
      __hip_atomic_fetch_add(bar, 1u, __ATOMIC_RELEASE, __HIP_MEMORY_SCOPE_AGENT);
      unsigned target = (unsigned)(t + 1) * NBLK_SCAN;
      while (__hip_atomic_load(bar, __ATOMIC_ACQUIRE, __HIP_MEMORY_SCOPE_AGENT) < target)
        __builtin_amdgcn_s_sleep(1);
    }
    __syncthreads();
    __threadfence();
  }
#pragma unroll
  for (int r = 0; r < 8; ++r) {
    int row = m0 + r + rshift;
    hsum[(size_t)row * HDIM + col] = hs[r];
  }
}

// ---------------- mean-pool (already summed) -> dense -> sigmoid ----------------
__global__ __launch_bounds__(128) void predict_kernel(
    const float* __restrict__ hsum, const float* __restrict__ Wd,
    const float* __restrict__ bd, float* __restrict__ out) {
  __shared__ float red[128];
  int b = blockIdx.x;
  float s = 0.0f;
  for (int c = threadIdx.x; c < HDIM; c += 128) s += hsum[(size_t)b * HDIM + c] * Wd[c];
  red[threadIdx.x] = s;
  __syncthreads();
  for (int st = 64; st > 0; st >>= 1) {
    if (threadIdx.x < st) red[threadIdx.x] += red[threadIdx.x + st];
    __syncthreads();
  }
  if (threadIdx.x == 0) {
    float m = red[0] * (1.0f / (float)TLEN);
    out[b] = sigm(m + bd[0]);
  }
}

extern "C" void kernel_launch(void* const* d_in, const int* in_sizes, int n_in,
                              void* d_out, int out_size, void* d_ws, size_t ws_size,
                              hipStream_t stream) {
  const float* essays  = (const float*)d_in[0];
  const float* W_lstm  = (const float*)d_in[1];
  const float* b_lstm  = (const float*)d_in[2];
  const float* W_dense = (const float*)d_in[3];
  const float* b_dense = (const float*)d_in[4];
  float* out = (float*)d_out;

  // workspace carve-up (all offsets 16B-aligned)
  char* ws = (char*)d_ws;
  size_t off = 0;
  float*  xproj = (float*)(ws + off);  off += (size_t)TLEN * BATCH * GDIM * 4;   // 512 MiB
  bf16_t* Ap    = (bf16_t*)(ws + off); off += (size_t)MROWS * DPAD * 2;          // 40 MiB
  bf16_t* BxP   = (bf16_t*)(ws + off); off += (size_t)KT_X * NTILES * 512 * 2;   // 1.31 MB
  bf16_t* BhP   = (bf16_t*)(ws + off); off += (size_t)KT_H * NTILES * 512 * 2;   // 2 MiB
  bf16_t* h0    = (bf16_t*)(ws + off); off += (size_t)BATCH * HDIM * 2;
  bf16_t* h1    = (bf16_t*)(ws + off); off += (size_t)BATCH * HDIM * 2;
  float*  hsum  = (float*)(ws + off);  off += (size_t)BATCH * HDIM * 4;
  unsigned* bar = (unsigned*)(ws + off);

  // 1. init state + barrier
  init_kernel<<<(BATCH * HDIM + 255) / 256, 256, 0, stream>>>(h0, bar);

  // 2. operand packing (bf16 conversion + WMMA-B tile swizzle)
  {
    size_t n = (size_t)MROWS * DPAD;
    pack_essays_kernel<<<(unsigned)((n + 255) / 256), 256, 0, stream>>>(essays, Ap);
  }
  {
    int n = (KT_X + KT_H) * NTILES * 64;
    pack_weights_kernel<<<(n + 255) / 256, 256, 0, stream>>>(W_lstm, BxP, BhP);
  }

  // 3. big parallel input-projection GEMM
  xproj_gemm_kernel<<<dim3(NTILES / 4, MROWS / 16), 128, 0, stream>>>(Ap, BxP, b_lstm, xproj);

  // 4. persistent recurrent scan (1024 steps, register-resident cell state)
  lstm_scan_kernel<<<NBLK_SCAN, 128, 0, stream>>>(xproj, BhP, h0, h1, hsum, bar);

  // 5. pool + dense + sigmoid
  predict_kernel<<<BATCH, 128, 0, stream>>>(hsum, W_dense, b_dense, out);
}
